// HL_LH_block_2_78151224918207
// MI455X (gfx1250) — matrix operand order repaired and checked by reading, not compiled, not run on hardware
//
#include <hip/hip_runtime.h>
#include <math.h>

// EGAT block for MI455X (gfx1250, wave32).
// - GEMMs: v_wmma_f32_16x16x32_f16, 32x128 block tile (2 M-subtiles per wave,
//   B fragment reused across both WMMAs), f32 inputs staged to f16 in LDS.
// - All [*,512] intermediates stored f16 -> halves dominant HBM traffic
//   (23.3 TB/s roofline; this workload is traffic/atomics bound, not FLOP bound).
// - Segment softmax: order-preserving u32-encoded atomicMax + f32 atomicAdd.
// - f_out [E,512] never materialized: reduced to logits [E,8] + head-sum [E,64]
//   in one pass. concat(edge[0::2],edge[1::2],axis=1) == reinterpret [E/2,128].

#define DF 64     // feature dim
#define NH 8      // heads
#define HD 512    // NH*DF

typedef __attribute__((ext_vector_type(16))) _Float16 v16h;
typedef __attribute__((ext_vector_type(8)))  float    v8f;
typedef __attribute__((ext_vector_type(2)))  _Float16 h2;
typedef __attribute__((ext_vector_type(2)))  float    f2;

__device__ __forceinline__ unsigned enc_f32(float x) {
  unsigned u = __float_as_uint(x);
  return (u & 0x80000000u) ? ~u : (u | 0x80000000u);
}
__device__ __forceinline__ float dec_f32(unsigned k) {
  unsigned u = (k & 0x80000000u) ? (k & 0x7FFFFFFFu) : ~k;
  return __uint_as_float(u);
}

__global__ void fill_zero_kernel(float* __restrict__ p, int n) {
  int i = blockIdx.x * blockDim.x + threadIdx.x;
  if (i < n) p[i] = 0.f;
}

// ---------------------------------------------------------------------------
// out[M,Nout] = X[M,K] @ W[Nout,K]^T (+ bias[Nout] if non-null)
// K in {64,128} (multiple of 32), Nout multiple of 16. OutT in {float, _Float16}.
// Block = 256 threads = 8 waves; block tile = 32 rows x 128 cols.
// Each wave: one 16-col slice, two 16-row subtiles sharing the B fragment.
// ---------------------------------------------------------------------------
template <typename OutT>
__global__ __launch_bounds__(256)
void gemm_xwt_wmma(const float* __restrict__ X, const float* __restrict__ W,
                   const float* __restrict__ bias, OutT* __restrict__ out,
                   int M, int K, int Nout)
{
  __shared__ _Float16 As[32 * 128];    // A tile (32 rows x K), f16
  __shared__ _Float16 Bs[128 * 128];   // W tile (128 cols x K), f16

  const int tid   = threadIdx.x;
  const int m0    = blockIdx.x * 32;
  const int nbase = blockIdx.y * 128;

  // stage A tile (f32 -> f16)
  for (int i = tid; i < 32 * K; i += 256) {
    int r = i / K, k = i - r * K;
    int m = m0 + r;
    As[r * K + k] = (m < M) ? (_Float16)X[(size_t)m * K + k] : (_Float16)0.f;
  }
  // stage W tile (f32 -> f16); B[k][n] == W[n][k]
  int ncols = Nout - nbase; if (ncols > 128) ncols = 128;
  for (int i = tid; i < ncols * K; i += 256) {
    int c = i / K, k = i - c * K;
    Bs[c * K + k] = (_Float16)W[(size_t)(nbase + c) * K + k];
  }
  __syncthreads();

  const int wave = tid >> 5;
  const int lane = tid & 31;
  const int n0   = nbase + wave * 16;
  if (n0 >= Nout) return;                 // idle waves when Nout < 128

  const int row = lane & 15;              // M-row (A) / N-col (B) within tile
  const int hi  = lane >> 4;              // lane half select

  v8f acc0 = {}, acc1 = {};
  for (int kk = 0; kk < K; kk += 32) {
    // B fragment: lanes 0-15 -> K {kk+0..7, kk+16..23}; lanes 16-31 -> {kk+8..15, kk+24..31} (col = n0+row)
    v16h bf;
    const _Float16* bp = &Bs[(wave * 16 + row) * K + kk + hi * 8];
#pragma unroll
    for (int i = 0; i < 8; ++i) { bf[i] = bp[i]; bf[8 + i] = bp[16 + i]; }

    // A fragments for the two 16-row subtiles:
    // lanes 0-15 -> K {kk+0..7, kk+16..23}; lanes 16-31 -> {kk+8..15, kk+24..31}
    v16h af0, af1;
    const _Float16* a0 = &As[row * K        + kk + hi * 8];
    const _Float16* a1 = &As[(16 + row) * K + kk + hi * 8];
#pragma unroll
    for (int i = 0; i < 8; ++i) {
      af0[i] = a0[i]; af0[8 + i] = a0[16 + i];
      af1[i] = a1[i]; af1[8 + i] = a1[16 + i];
    }
    acc0 = __builtin_amdgcn_wmma_f32_16x16x32_f16(false, af0, false, bf,
                                                  (short)0, acc0, false, false);
    acc1 = __builtin_amdgcn_wmma_f32_16x16x32_f16(false, af1, false, bf,
                                                  (short)0, acc1, false, false);
  }

  // C layout: VGPR r -> M = r (lanes 0-15) / r+8 (lanes 16-31); lane&15 -> N
  const int n  = n0 + row;
  const float bv = bias ? bias[n] : 0.f;
#pragma unroll
  for (int r = 0; r < 8; ++r) {
    int m = m0 + r + hi * 8;
    if (m < M)      out[(size_t)m * Nout + n]        = (OutT)(acc0[r] + bv);
    if (m + 16 < M) out[(size_t)(m + 16) * Nout + n] = (OutT)(acc1[r] + bv);
  }
}

// ---------------------------------------------------------------------------
// Edge pass 1: one wave per edge. Lane L owns d = 2L, 2L+1 (coalesced h2 loads).
//   f = lrelu(fni[src] + fnj[dst] + ffij[e])     (bias fused into ffij)
//   edge_out[e,:] = sum over heads of f          ([E,64], f32)
//   logits[e,h]   = <f[h,:], attn[h,:]>          ([E,8])
//   mEnc[dst,h]   = atomicMax(enc(logit))
// ---------------------------------------------------------------------------
__global__ __launch_bounds__(256)
void edge_logits_kernel(const _Float16* __restrict__ fni, const _Float16* __restrict__ fnj,
                        const _Float16* __restrict__ ffij, const float* __restrict__ attn,
                        const int* __restrict__ src, const int* __restrict__ dst,
                        float* __restrict__ edge_out, float* __restrict__ logits,
                        unsigned* __restrict__ mEnc, int E)
{
  int e    = blockIdx.x * 8 + (threadIdx.x >> 5);
  int lane = threadIdx.x & 31;
  if (e >= E) return;

  const _Float16* fi = fni  + (size_t)src[e] * HD;
  const _Float16* fj = fnj  + (size_t)dst[e] * HD;
  const _Float16* fe = ffij + (size_t)e      * HD;

  float sum0 = 0.f, sum1 = 0.f;
  float lg[NH];
#pragma unroll
  for (int h = 0; h < NH; ++h) lg[h] = 0.f;

#pragma unroll
  for (int t = 0; t < NH; ++t) {               // t = head; d = 2*lane, 2*lane+1
    int j = 2 * lane + DF * t;
    h2 a = *(const h2*)(fi + j);
    h2 b = *(const h2*)(fj + j);
    h2 c = *(const h2*)(fe + j);
    float f0 = (float)a.x + (float)b.x + (float)c.x;
    float f1 = (float)a.y + (float)b.y + (float)c.y;
    f0 = (f0 > 0.f) ? f0 : 0.01f * f0;         // leaky_relu(0.01)
    f1 = (f1 > 0.f) ? f1 : 0.01f * f1;
    sum0 += f0; sum1 += f1;
    lg[t] += f0 * attn[t * DF + 2 * lane] + f1 * attn[t * DF + 2 * lane + 1];
  }
  f2 eo; eo.x = sum0; eo.y = sum1;
  *(f2*)(edge_out + (size_t)e * DF + 2 * lane) = eo;

  const int dn = dst[e];
#pragma unroll
  for (int h = 0; h < NH; ++h) {
    float v = lg[h];
    v += __shfl_xor(v, 16, 32);
    v += __shfl_xor(v, 8, 32);
    v += __shfl_xor(v, 4, 32);
    v += __shfl_xor(v, 2, 32);
    v += __shfl_xor(v, 1, 32);
    if (lane == 0) {
      logits[(size_t)e * NH + h] = v;
      atomicMax(&mEnc[(size_t)dn * NH + h], enc_f32(v));
    }
  }
}

// Edge pass 2: a = exp(logit - m[dst]); z[dst] += a  (one thread per (e,h))
__global__ void edge_exp_kernel(const float* __restrict__ logits,
                                const int* __restrict__ dst,
                                const unsigned* __restrict__ mEnc,
                                float* __restrict__ a, float* __restrict__ z, int E)
{
  int i = blockIdx.x * blockDim.x + threadIdx.x;
  if (i >= E * NH) return;
  int e = i >> 3, h = i & 7;
  int dn = dst[e];
  float av = __expf(logits[i] - dec_f32(mEnc[(size_t)dn * NH + h]));
  a[i] = av;
  atomicAdd(&z[(size_t)dn * NH + h], av);
}

// Edge pass 3: node_out[dst,d] += sum_h hn[src,h,d] * (a[e,h]/z[dst,h])
// One wave per edge; lane L owns d = 2L, 2L+1. 64 f32 atomics/edge.
__global__ __launch_bounds__(256)
void edge_scatter_kernel(const _Float16* __restrict__ hn, const float* __restrict__ a,
                         const float* __restrict__ z, const int* __restrict__ src,
                         const int* __restrict__ dst, float* __restrict__ node_out, int E)
{
  int e    = blockIdx.x * 8 + (threadIdx.x >> 5);
  int lane = threadIdx.x & 31;
  if (e >= E) return;
  const int sn = src[e], dn = dst[e];

  float alpha[NH];
#pragma unroll
  for (int h = 0; h < NH; ++h)
    alpha[h] = a[(size_t)e * NH + h] / z[(size_t)dn * NH + h];

  const _Float16* hp = hn + (size_t)sn * HD;
  float v0 = 0.f, v1 = 0.f;
#pragma unroll
  for (int h = 0; h < NH; ++h) {
    h2 v = *(const h2*)(hp + h * DF + 2 * lane);
    v0 += (float)v.x * alpha[h];
    v1 += (float)v.y * alpha[h];
  }
  atomicAdd(&node_out[(size_t)dn * DF + 2 * lane],     v0);
  atomicAdd(&node_out[(size_t)dn * DF + 2 * lane + 1], v1);
}

// ---------------------------------------------------------------------------
// Host-side orchestration
// ---------------------------------------------------------------------------
struct EgatP { const float *Wn, *bn, *Wni, *Wnj, *Wf, *attn, *bias; };

template <typename OutT>
static void run_gemm(const float* X, const float* W, const float* bias, OutT* out,
                     int M, int K, int Nout, hipStream_t s)
{
  dim3 g((M + 31) / 32, (Nout + 127) / 128);
  gemm_xwt_wmma<OutT><<<g, 256, 0, s>>>(X, W, bias, out, M, K, Nout);
}

static void run_egat(int N, int E, const int* src, const int* dst,
                     const float* xn, const float* xe, const EgatP& p,
                     _Float16* fni, _Float16* fnj, _Float16* hn, _Float16* ffij,
                     float* logits, float* a, unsigned* mEnc, float* z,
                     float* node_out, float* edge_out, hipStream_t s)
{
  run_gemm(xn, p.Wni, (const float*)nullptr, fni, N, DF, HD, s);
  run_gemm(xn, p.Wnj, (const float*)nullptr, fnj, N, DF, HD, s);
  run_gemm(xe, p.Wf,  p.bias,                ffij, E, DF, HD, s);  // edge bias fused
  run_gemm(xn, p.Wn,  p.bn,                  hn,   N, DF, HD, s);

  int nzh = N * NH;
  fill_zero_kernel<<<(nzh + 255) / 256, 256, 0, s>>>((float*)mEnc, nzh); // enc 0 < enc(any)
  fill_zero_kernel<<<(nzh + 255) / 256, 256, 0, s>>>(z, nzh);
  int nzd = N * DF;
  fill_zero_kernel<<<(nzd + 255) / 256, 256, 0, s>>>(node_out, nzd);

  dim3 ge((E + 7) / 8);
  edge_logits_kernel<<<ge, 256, 0, s>>>(fni, fnj, ffij, p.attn, src, dst,
                                        edge_out, logits, mEnc, E);
  edge_exp_kernel<<<((E * NH) + 255) / 256, 256, 0, s>>>(logits, dst, mEnc, a, z, E);
  edge_scatter_kernel<<<ge, 256, 0, s>>>(hn, a, z, src, dst, node_out, E);
}

extern "C" void kernel_launch(void* const* d_in, const int* in_sizes, int n_in,
                              void* d_out, int out_size, void* d_ws, size_t ws_size,
                              hipStream_t stream)
{
  const int N1 = 10000, E1 = 200000, N2 = 100000, E2 = 200000;

  const float* node_feats = (const float*)d_in[0];
  const float* edge_feats = (const float*)d_in[1];
  const float* edge_path  = (const float*)d_in[2];
  const int*   src1       = (const int*)d_in[3];
  const int*   dst1       = (const int*)d_in[4];
  const int*   src2       = (const int*)d_in[5];
  const int*   dst2       = (const int*)d_in[6];

  auto params = [&](int b) {
    EgatP p;
    p.Wn   = (const float*)d_in[b + 0];
    p.bn   = (const float*)d_in[b + 1];
    p.Wni  = (const float*)d_in[b + 2];
    p.Wnj  = (const float*)d_in[b + 3];
    p.Wf   = (const float*)d_in[b + 4];
    p.attn = (const float*)d_in[b + 5];
    p.bias = (const float*)d_in[b + 6];
    return p;
  };
  EgatP p3a = params(7), p3b = params(14), p4 = params(21);
  const float* lin_W = (const float*)d_in[28];  // [64, 128]
  const float* lin_b = (const float*)d_in[29];  // [64]

  // workspace carve (sized for the largest layer; buffers reused per layer)
  char* wp = (char*)d_ws;
  auto take = [&](size_t bytes) -> void* {
    void* r = (void*)wp;
    wp += (bytes + 255) & ~(size_t)255;
    return r;
  };
  _Float16* fni    = (_Float16*)take((size_t)N2 * HD * 2);
  _Float16* fnj    = (_Float16*)take((size_t)N2 * HD * 2);
  _Float16* hn     = (_Float16*)take((size_t)N2 * HD * 2);
  _Float16* ffij   = (_Float16*)take((size_t)E1 * HD * 2);
  float*    logits = (float*)take((size_t)E1 * NH * 4);
  float*    aw     = (float*)take((size_t)E1 * NH * 4);
  unsigned* mEnc   = (unsigned*)take((size_t)N2 * NH * 4);
  float*    zw     = (float*)take((size_t)N2 * NH * 4);
  float*    nodeA  = (float*)take((size_t)N2 * DF * 4);
  float*    nodeB  = (float*)take((size_t)N2 * DF * 4);
  float*    edgeA  = (float*)take((size_t)E1 * DF * 4);
  float*    edgeB  = (float*)take((size_t)E1 * DF * 4);
  (void)ws_size; (void)in_sizes; (void)n_in; (void)out_size;

  // gcn_3 layer a: (node_feats, edge_feats) -> (nodeA, edgeA)
  run_egat(N1, E1, src1, dst1, node_feats, edge_feats, p3a,
           fni, fnj, hn, ffij, logits, aw, mEnc, zw, nodeA, edgeA, stream);
  // gcn_3 layer b: -> (nodeB, edgeB)
  run_egat(N1, E1, src1, dst1, nodeA, edgeA, p3b,
           fni, fnj, hn, ffij, logits, aw, mEnc, zw, nodeB, edgeB, stream);

  // concat(edge[0::2], edge[1::2], axis=1) on row-major [E1,64] == view [E1/2,128]
  // node_path = x11 @ lin_W^T + lin_b  -> reuse nodeA as [N2,64] (f32 output)
  run_gemm(edgeB, lin_W, lin_b, nodeA, N2, 2 * DF, DF, stream);

  // gcn_4: outputs straight into d_out (node part, then edge part)
  float* out_node = (float*)d_out;
  float* out_edge = (float*)d_out + (size_t)N2 * DF;
  run_egat(N2, E2, src2, dst2, nodeA, edge_path, p4,
           fni, fnj, hn, ffij, logits, aw, mEnc, zw, out_node, out_edge, stream);
}
